// Toeplizt_76106820485416
// MI455X (gfx1250) — compile-verified, matrix operand order checked by reading, not run on hardware
//
#include <hip/hip_runtime.h>
#include <hip/hip_bf16.h>

// Toeplitz matvec as WMMA GEMM on gfx1250 (wave32).
// out[b,m,e] = sum_k T[m,k] * x[b,k,e],  T 2048x2048 shared across b,e.
//   T[m,m]=zero; T[m,k]=pos[m-k-1] (m>k); T[m,k]=neg[(N-1)-(k-m)] (m<k).
//
// Both operands are pre-packed in workspace in the exact 16-bit WMMA fragment
// layout (lane l<16: K in {0..7,16..23}; lane l>=16: K in {8..15,24..31}),
// 32 contiguous bytes per lane-slot, so GEMM staging is a pure linear
// GLOBAL_LOAD_ASYNC_TO_LDS_B128 stream tracked by ASYNCcnt.

typedef __attribute__((ext_vector_type(16))) _Float16 v16h;
typedef __attribute__((ext_vector_type(8)))  float    v8f;
typedef __attribute__((ext_vector_type(4)))  int      v4i;

#define NN 2048   // sequence length (M and K of the GEMM)
#define EE 1024   // embedding
#define BB 8      // batch

#if __has_builtin(__builtin_amdgcn_global_load_async_to_lds_b128)
// builtin signature (from hipcc diagnostics): (v4i* src, v4i* lds_dst, imm, imm)
#define ASYNC_COPY32(gp, lp)                                                   \
  do {                                                                         \
    __builtin_amdgcn_global_load_async_to_lds_b128((v4i*)(gp), (v4i*)(lp), 0,  \
                                                   0);                         \
    __builtin_amdgcn_global_load_async_to_lds_b128((v4i*)(gp), (v4i*)(lp), 16, \
                                                   0);                         \
  } while (0)
#if __has_builtin(__builtin_amdgcn_s_wait_asynccnt)
#define ASYNC_WAIT() __builtin_amdgcn_s_wait_asynccnt(0)
#else
#define ASYNC_WAIT() asm volatile("s_wait_asynccnt 0x0" ::: "memory")
#endif
#else  // fallback: classic reg-staged copy
#define ASYNC_COPY32(gp, lp)                                                   \
  do {                                                                         \
    v4i t0_ = *(const v4i*)(gp);                                               \
    v4i t1_ = *(const v4i*)((const _Float16*)(gp) + 8);                        \
    *(v4i*)(lp) = t0_;                                                         \
    *(v4i*)((_Float16*)(lp) + 8) = t1_;                                        \
  } while (0)
#define ASYNC_WAIT()
#endif

// fragment-local K index for half h (0..15) in lane-group g (0..1)
__device__ __forceinline__ int kloc_of(int h, int g) {
  return (h & 7) | ((h >> 3) << 4) | (g << 3);
}

// ---------------------------------------------------------------------------
// Prep 1: Toeplitz matrix, written directly in packed fragment layout.
// Tp half-index: ((mf*64 + kk)*32 + lane)*16 + h
//   m = mf*16 + (lane&15),  k = kk*32 + kloc(h, lane>>4)
// ---------------------------------------------------------------------------
__global__ __launch_bounds__(256) void build_T(const float* __restrict__ pos,
                                               const float* __restrict__ zero,
                                               const float* __restrict__ neg,
                                               _Float16* __restrict__ Tp) {
  const int gid  = blockIdx.x * 256 + threadIdx.x;   // 0 .. NN*NN-1
  const int h    = gid & 15;
  const int lane = (gid >> 4) & 31;
  const int kk   = (gid >> 9) & 63;
  const int mf   = gid >> 15;                        // 0..127
  const int m = mf * 16 + (lane & 15);
  const int k = kk * 32 + kloc_of(h, lane >> 4);
  float v;
  if (m == k)      v = zero[0];
  else if (m > k)  v = pos[m - k - 1];
  else             v = neg[(NN - 1) - (k - m)];
  Tp[gid] = (_Float16)v;
}

// ---------------------------------------------------------------------------
// Prep 2: x[b,k,e] f32 -> packed f16 fragments.
// Xp half-index: (((b*64 + ef)*64 + kk)*32 + lane)*16 + h
//   e = ef*16 + (lane&15),  k = kk*32 + kloc(h, lane>>4)
// Per-wave 32(k) x 16(e) tile through LDS: fully-used 64B row reads,
// coalesced 1KB packed writes.
// ---------------------------------------------------------------------------
__global__ __launch_bounds__(256) void xpose(const float* __restrict__ x,
                                             _Float16* __restrict__ Xp) {
  __shared__ _Float16 lt[8][512];                    // [wave][k_row*16 + e_loc]
  const int lane = threadIdx.x & 31;
  const int wave = threadIdx.x >> 5;
  const int tile = blockIdx.x * 8 + wave;            // (b*64 + ef)*64 + kk
  const int kk = tile & 63;
  const int ef = (tile >> 6) & 63;
  const int b  = tile >> 12;

  // read one k-row (16 consecutive e) -> one full 64B line per thread
  const float* src = x + ((size_t)b * NN + kk * 32 + lane) * EE + ef * 16;
  _Float16* row = &lt[wave][lane * 16];
#pragma unroll
  for (int j = 0; j < 16; j += 4) {
    float4 r = *(const float4*)(src + j);
    row[j + 0] = (_Float16)r.x;
    row[j + 1] = (_Float16)r.y;
    row[j + 2] = (_Float16)r.z;
    row[j + 3] = (_Float16)r.w;
  }
  __syncthreads();

  // gather this lane's packed fragment slot and write 32B coalesced
  union { _Float16 h[16]; v4i q[2]; } u;
#pragma unroll
  for (int h = 0; h < 16; ++h)
    u.h[h] = lt[wave][kloc_of(h, lane >> 4) * 16 + (lane & 15)];
  _Float16* dst = Xp + (size_t)tile * 512 + lane * 16;
  *(v4i*)(dst)     = u.q[0];
  *(v4i*)(dst + 8) = u.q[1];
}

// ---------------------------------------------------------------------------
// GEMM: block = 256 threads = 8 waves (4 x 2 wave grid).
// Block tile: 256(M) x 128(E), K-step 32, double-buffered LDS (48 KB).
// Wave tile: 64 x 64 = 4x4 fragments -> 16 WMMAs per K-step (1:1 with
// ds_load_b128). Staging: 6 async b128 copies per thread per K-step.
// ---------------------------------------------------------------------------
__global__ __launch_bounds__(256) void toeplitz_wmma(
    const _Float16* __restrict__ Tp,   // packed [128][64][32][16]
    const _Float16* __restrict__ Xp,   // packed [BB][64][64][32][16]
    float* __restrict__ out) {         // [BB][NN][EE] f32
  __shared__ __align__(32) _Float16 smA[2][8192];    // 16 frags x 512 halves
  __shared__ __align__(32) _Float16 smB[2][4096];    //  8 frags x 512 halves

  const int tid  = threadIdx.x;
  const int lane = tid & 31;
  const int wave = tid >> 5;
  const int wm   = wave >> 1;                        // 0..3 (M)
  const int wn   = wave & 1;                         // 0..1 (E)

  const int blockE = blockIdx.x * 128;
  const int blockM = blockIdx.y * 256;
  const int b      = blockIdx.z;

  // staging sources: wave stages A frags {2w, 2w+1} and B frag {w}
  const int fA0 = 2 * wave, fA1 = 2 * wave + 1;
  const _Float16* gA0 = Tp + (((size_t)(blockM / 16 + fA0) * 64) * 32 + lane) * 16;
  const _Float16* gA1 = Tp + (((size_t)(blockM / 16 + fA1) * 64) * 32 + lane) * 16;
  const _Float16* gB  = Xp + ((((size_t)b * 64 + blockE / 16 + wave) * 64) * 32 + lane) * 16;

  v8f acc[4][4] = {};

  // stage slab 0 into buffer 0
  ASYNC_COPY32(gA0, &smA[0][fA0 * 512 + lane * 16]);
  ASYNC_COPY32(gA1, &smA[0][fA1 * 512 + lane * 16]);
  ASYNC_COPY32(gB,  &smB[0][wave * 512 + lane * 16]);
  ASYNC_WAIT();
  __syncthreads();

  for (int kk = 0; kk < NN / 32; ++kk) {
    const int cur = kk & 1;
    const int nxt = cur ^ 1;

    if (kk < NN / 32 - 1) {                          // async-stage next slab
      const size_t adv = (size_t)(kk + 1) * 512;
      ASYNC_COPY32(gA0 + adv, &smA[nxt][fA0 * 512 + lane * 16]);
      ASYNC_COPY32(gA1 + adv, &smA[nxt][fA1 * 512 + lane * 16]);
      ASYNC_COPY32(gB  + adv, &smB[nxt][wave * 512 + lane * 16]);
    }

    v16h af[4], bf[4];
#pragma unroll
    for (int fm = 0; fm < 4; ++fm)
      af[fm] = *(const v16h*)&smA[cur][(wm * 4 + fm) * 512 + lane * 16];
#pragma unroll
    for (int fn = 0; fn < 4; ++fn)
      bf[fn] = *(const v16h*)&smB[cur][(wn * 4 + fn) * 512 + lane * 16];

#pragma unroll
    for (int fm = 0; fm < 4; ++fm)
#pragma unroll
      for (int fn = 0; fn < 4; ++fn)
        acc[fm][fn] = __builtin_amdgcn_wmma_f32_16x16x32_f16(
            false, af[fm], false, bf[fn], (short)0, acc[fm][fn], false, false);

    ASYNC_WAIT();
    __syncthreads();
  }

  // ---- epilogue: C/D layout -> out[b, m, e] (f32)
  //   VGPR r : lanes 0-15 -> M=r, lanes 16-31 -> M=r+8 ; N = lane & 15.
#pragma unroll
  for (int fm = 0; fm < 4; ++fm) {
    const int mrow = blockM + wm * 64 + fm * 16 + ((lane >> 4) << 3);
#pragma unroll
    for (int fn = 0; fn < 4; ++fn) {
      const int ecol = blockE + wn * 64 + fn * 16 + (lane & 15);
      float* op = out + ((size_t)b * NN + mrow) * EE + ecol;
#pragma unroll
      for (int r = 0; r < 8; ++r)
        op[(size_t)r * EE] = acc[fm][fn][r];
    }
  }
}

// ---------------------------------------------------------------------------
extern "C" void kernel_launch(void* const* d_in, const int* in_sizes, int n_in,
                              void* d_out, int out_size, void* d_ws, size_t ws_size,
                              hipStream_t stream) {
  const float* x    = (const float*)d_in[0];  // [B,N,E] f32
  const float* pos  = (const float*)d_in[1];  // [N-1]
  const float* zero = (const float*)d_in[2];  // [1]
  const float* neg  = (const float*)d_in[3];  // [N-1]
  float* out = (float*)d_out;                 // [B,N,E] f32

  char* ws = (char*)d_ws;
  _Float16* Tp = (_Float16*)ws;                                        // 8 MB
  _Float16* Xp = (_Float16*)(ws + (size_t)NN * NN * sizeof(_Float16)); // 32 MB

  // 1) Toeplitz matrix -> packed f16 fragments
  build_T<<<(NN * NN) / 256, 256, 0, stream>>>(pos, zero, neg, Tp);

  // 2) x -> packed f16 fragments (transpose through LDS)
  xpose<<<(BB * EE / 16) * (NN / 32) / 8, 256, 0, stream>>>(x, Xp);

  // 3) WMMA GEMM: grid (E/128, N/256, B)
  dim3 grid(EE / 128, NN / 256, BB);
  toeplitz_wmma<<<grid, 256, 0, stream>>>(Tp, Xp, out);
}
